// PagedAttention_77034533421648
// MI455X (gfx1250) — matrix operand order repaired and checked by reading, not compiled, not run on hardware
//
#include <hip/hip_runtime.h>
#include <hip/hip_bf16.h>

// Paged attention decode, flash-style single pass, f16 WMMA on gfx1250.
#define NUM_HEADS_  32
#define HEAD_SIZE_  128
#define NUM_KV_     8
#define GQ_         4        // NUM_HEADS / NUM_KV (GQA group)
#define BS_         16       // page (block) size in tokens
#define NUM_SEQS_   64
#define MAX_BLOCKS_ 64
#define PAGE_FLTS_  (HEAD_SIZE_ * BS_)             // 2048 floats per (block,kv-head) page
#define BLK_STRIDE_ ((size_t)NUM_KV_ * PAGE_FLTS_) // floats between physical blocks
#define SCALE_      0.08838834764831845f

typedef __attribute__((ext_vector_type(16))) _Float16 v16h;
typedef __attribute__((ext_vector_type(8)))  float    v8f;

// A-matrix (16x32 f16) element->K mapping (contiguous, matches B):
// element e of lane half hi: K = hi*16 + e
__device__ __forceinline__ int kpatA(int e, int hi) {
    return hi * 16 + e;
}
// B-matrix (32x16 f16): lane%16 = N; element e holds K = hi*16 + e.

__device__ __forceinline__ v16h pack16(float4 a0, float4 a1, float4 a2, float4 a3) {
    v16h r;
    r[0]=(_Float16)a0.x;  r[1]=(_Float16)a0.y;  r[2]=(_Float16)a0.z;  r[3]=(_Float16)a0.w;
    r[4]=(_Float16)a1.x;  r[5]=(_Float16)a1.y;  r[6]=(_Float16)a1.z;  r[7]=(_Float16)a1.w;
    r[8]=(_Float16)a2.x;  r[9]=(_Float16)a2.y;  r[10]=(_Float16)a2.z; r[11]=(_Float16)a2.w;
    r[12]=(_Float16)a3.x; r[13]=(_Float16)a3.y; r[14]=(_Float16)a3.z; r[15]=(_Float16)a3.w;
    return r;
}

// One 32-token tile. TAIL=true only for the single tile containing token ctx-1:
// that tile needs the length mask and the knew/vnew substitution; all other
// tiles take the clean all-cache b128 load path.
template <bool TAIL>
__device__ __forceinline__ void attn_tile(
    int base, int blk0, int blk1, int lo, int hi, int ctx, int last,
    const float* __restrict__ kch,       // kc + h page offset
    const float* __restrict__ vch,       // vc + h page offset
    const float* __restrict__ krow_new,
    const float* __restrict__ vrow_new,
    const v16h* __restrict__ aq,         // [4] Q A-tiles
    v8f* __restrict__ o,                 // [8] running O accumulators
    float* __restrict__ mrun,            // [GQ_]
    float* __restrict__ lrun,            // [GQ_]
    float* __restrict__ pwave)           // per-wave LDS prob buffer (128 floats)
{
    const float* kp0 = kch + (size_t)blk0 * BLK_STRIDE_;
    const float* kp1 = kch + (size_t)blk1 * BLK_STRIDE_;
    const int tok0 = base + lo;          // subtile-0 token (N col of this lane)
    const int tok1 = base + 16 + lo;     // subtile-1 token

    // -------- scores: D = Q x K^T, both 16-token subtiles interleaved --------
    // Interleaving gives consecutive WMMAs independent B/C operands, so the
    // f16-WMMA->VALU hazard slots are filled with the other subtile's cvts.
    v8f acc0 = v8f{}, acc1 = v8f{};
    #pragma unroll
    for (int c = 0; c < 4; ++c) {
        // dims d = c*32 + hi*16 + [0..15] live in X-groups g0,g0+1 of this token
        const int g0 = c * 4 + hi * 2;
        const float* kg0 = kp0 + (size_t)g0 * (BS_ * 8) + lo * 8;
        const float* kg1 = kp1 + (size_t)g0 * (BS_ * 8) + lo * 8;
        float4 a0 = *(const float4*)(kg0);
        float4 a1 = *(const float4*)(kg0 + 4);
        float4 a2 = *(const float4*)(kg0 + BS_ * 8);
        float4 a3 = *(const float4*)(kg0 + BS_ * 8 + 4);
        float4 b0 = *(const float4*)(kg1);
        float4 b1 = *(const float4*)(kg1 + 4);
        float4 b2 = *(const float4*)(kg1 + BS_ * 8);
        float4 b3 = *(const float4*)(kg1 + BS_ * 8 + 4);
        if (TAIL) {                                  // new token's K from knew
            const float* kn = krow_new + c * 32 + hi * 16;
            if (tok0 == last) {
                a0 = *(const float4*)(kn);     a1 = *(const float4*)(kn + 4);
                a2 = *(const float4*)(kn + 8); a3 = *(const float4*)(kn + 12);
            }
            if (tok1 == last) {
                b0 = *(const float4*)(kn);     b1 = *(const float4*)(kn + 4);
                b2 = *(const float4*)(kn + 8); b3 = *(const float4*)(kn + 12);
            }
        }
        v16h bk0 = pack16(a0, a1, a2, a3);
        v16h bk1 = pack16(b0, b1, b2, b3);
        acc0 = __builtin_amdgcn_wmma_f32_16x16x32_f16(
                   false, aq[c], false, bk0, (short)0, acc0, false, false);
        acc1 = __builtin_amdgcn_wmma_f32_16x16x32_f16(
                   false, aq[c], false, bk1, (short)0, acc1, false, false);
    }
    #pragma unroll
    for (int r = 0; r < 8; ++r) {
        float s0 = acc0[r] * SCALE_;
        float s1 = acc1[r] * SCALE_;
        acc0[r] = (!TAIL || tok0 < ctx) ? s0 : -1e30f;
        acc1[r] = (!TAIL || tok1 < ctx) ? s1 : -1e30f;
    }

    // -------- online softmax, rows 0..3 (valid on lanes 0-15) --------
    float fac[GQ_];
    #pragma unroll
    for (int g = 0; g < GQ_; ++g) {
        float t = fmaxf(acc0[g], acc1[g]);
        #pragma unroll
        for (int off = 8; off >= 1; off >>= 1)
            t = fmaxf(t, __shfl_xor(t, off, 16));
        float mnew = fmaxf(mrun[g], t);
        float f  = __expf(mrun[g] - mnew);
        float p0 = __expf(acc0[g] - mnew);
        float p1 = __expf(acc1[g] - mnew);
        float rs = p0 + p1;
        #pragma unroll
        for (int off = 8; off >= 1; off >>= 1)
            rs += __shfl_xor(rs, off, 16);
        mrun[g] = mnew;
        lrun[g] = lrun[g] * f + rs;
        fac[g]  = f;
        if (hi == 0) {                               // D-layout -> LDS (rows 0-3)
            pwave[g * 32 + lo]      = p0;
            pwave[g * 32 + 16 + lo] = p1;
        }
    }
    asm volatile("" ::: "memory");                   // order LDS store -> load

    // rescale running O (head rows 0-3 live on lanes 0-15, VGPR slots 0-3)
    #pragma unroll
    for (int c = 0; c < 8; ++c)
        #pragma unroll
        for (int g = 0; g < GQ_; ++g)
            o[c][g] *= fac[g];

    // -------- probs A-tile (16x32): lane m=lo, K=token via kpatA --------
    v16h pa;
    #pragma unroll
    for (int e = 0; e < 16; ++e) {
        float pv = pwave[(lo & 3) * 32 + kpatA(e, hi)];
        if (lo >= GQ_) pv = 0.0f;                    // padded head rows -> 0
        pa[e] = (_Float16)pv;
    }

    // -------- O += P x V : 8 head-dim chunks of 16 --------
    const float* vpage = (hi ? vch + (size_t)blk1 * BLK_STRIDE_
                             : vch + (size_t)blk0 * BLK_STRIDE_);
    const int le = TAIL ? (last - base - hi * 16) : -1;   // new token within this half
    v16h bvb[2];                                     // double-buffer: break WMMA WAR
    #pragma unroll
    for (int c = 0; c < 8; ++c) {
        const int d = c * 16 + lo;                   // head dim (N col) for this lane
        const float* vrow = vpage + (size_t)d * BS_; // 16 contiguous tokens
        float4 b0 = *(const float4*)(vrow);
        float4 b1 = *(const float4*)(vrow + 4);
        float4 b2 = *(const float4*)(vrow + 8);
        float4 b3 = *(const float4*)(vrow + 12);
        bvb[c & 1] = pack16(b0, b1, b2, b3);
        if (TAIL && (unsigned)le < 16u)
            bvb[c & 1][le] = (_Float16)vrow_new[d];  // new token's V from vnew
        o[c] = __builtin_amdgcn_wmma_f32_16x16x32_f16(
                   false, pa, false, bvb[c & 1], (short)0, o[c], false, false);
    }
}

__global__ __launch_bounds__(128)
void PagedAttention_77034533421648_kernel(const float* __restrict__ q,
                                          const float* __restrict__ knew,
                                          const float* __restrict__ vnew,
                                          const float* __restrict__ kc,
                                          const float* __restrict__ vc,
                                          const int*  __restrict__ bt,
                                          const int*  __restrict__ ctxlen,
                                          float*      __restrict__ out) {
    __shared__ float pbuf[4][GQ_ * 32];            // per-wave prob transpose buffer
    __shared__ float obuf[4][GQ_][HEAD_SIZE_];     // per-wave unnormalized O
    __shared__ float mstat[4][GQ_], lstat[4][GQ_]; // per-wave flash stats

    const int h    = blockIdx.x;        // kv head
    const int s    = blockIdx.y;        // sequence
    const int wave = threadIdx.x >> 5;
    const int lane = threadIdx.x & 31;
    const int lo   = lane & 15;
    const int hi   = lane >> 4;

    const int ctx  = ctxlen[s];
    const int last = ctx - 1;           // token whose K/V come from knew/vnew
    const int* btab = bt + s * MAX_BLOCKS_;

    const float* kch = kc + (size_t)h * PAGE_FLTS_;
    const float* vch = vc + (size_t)h * PAGE_FLTS_;
    const float* krow_new = knew + ((size_t)s * NUM_KV_ + h) * HEAD_SIZE_;
    const float* vrow_new = vnew + ((size_t)s * NUM_KV_ + h) * HEAD_SIZE_;

    // ---- Q A-tiles: 16(heads, 4 real) x 32(dim chunk), 4 chunks of head dim ----
    v16h aq[4];
    const float* qrow = q + ((size_t)s * NUM_HEADS_ + h * GQ_ + (lo & 3)) * HEAD_SIZE_;
    #pragma unroll
    for (int c = 0; c < 4; ++c) {
        #pragma unroll
        for (int e = 0; e < 16; ++e) {
            float val = (lo < GQ_) ? qrow[c * 32 + kpatA(e, hi)] : 0.0f;
            aq[c][e] = (_Float16)val;
        }
    }

    v8f o[8];
    #pragma unroll
    for (int c = 0; c < 8; ++c) o[c] = v8f{};
    float mrun[GQ_], lrun[GQ_];
    #pragma unroll
    for (int g = 0; g < GQ_; ++g) { mrun[g] = -1e30f; lrun[g] = 0.0f; }

    const int ntiles   = (ctx + 31) >> 5;          // 32-token tiles
    const int lastTile = ntiles - 1;               // the only tile needing mask+subst
    for (int tile = wave; tile < ntiles; tile += 4) {
        const int base = tile << 5;
        const int blk0 = btab[(base >> 4) + 0];
        const int blk1 = btab[(base >> 4) + 1];

        if (tile + 4 < ntiles) {                   // prefetch next K page
            int nb = btab[(base + 128) >> 4];
            __builtin_prefetch(kch + (size_t)nb * BLK_STRIDE_, 0, 1);
        }

        if (tile < lastTile)                       // wave-uniform scalar branch
            attn_tile<false>(base, blk0, blk1, lo, hi, ctx, last,
                             kch, vch, krow_new, vrow_new,
                             aq, o, mrun, lrun, &pbuf[wave][0]);
        else
            attn_tile<true>(base, blk0, blk1, lo, hi, ctx, last,
                            kch, vch, krow_new, vrow_new,
                            aq, o, mrun, lrun, &pbuf[wave][0]);
    }

    // -------- cross-wave merge via LDS log-sum-exp --------
    if (hi == 0) {
        #pragma unroll
        for (int c = 0; c < 8; ++c)
            #pragma unroll
            for (int g = 0; g < GQ_; ++g)
                obuf[wave][g][c * 16 + lo] = o[c][g];
        if (lo == 0) {
            #pragma unroll
            for (int g = 0; g < GQ_; ++g) { mstat[wave][g] = mrun[g]; lstat[wave][g] = lrun[g]; }
        }
    }
    __syncthreads();

    for (int i = threadIdx.x; i < GQ_ * HEAD_SIZE_; i += 128) {
        int g = i >> 7;
        int d = i & 127;
        float M = -1e30f;
        #pragma unroll
        for (int w = 0; w < 4; ++w) M = fmaxf(M, mstat[w][g]);
        float L = 0.0f, val = 0.0f;
        #pragma unroll
        for (int w = 0; w < 4; ++w) {
            float f = __expf(mstat[w][g] - M);
            L   += lstat[w][g] * f;
            val += obuf[w][g][d] * f;
        }
        out[((size_t)s * NUM_HEADS_ + h * GQ_ + g) * HEAD_SIZE_ + d] = val / L;
    }
}

extern "C" void kernel_launch(void* const* d_in, const int* in_sizes, int n_in,
                              void* d_out, int out_size, void* d_ws, size_t ws_size,
                              hipStream_t stream) {
    const float* q    = (const float*)d_in[0];  // [64, 32*128]
    const float* knew = (const float*)d_in[1];  // [64, 8*128]
    const float* vnew = (const float*)d_in[2];  // [64, 8*128]
    const float* kc   = (const float*)d_in[3];  // [4096, 8, 16, 16, 8]
    const float* vc   = (const float*)d_in[4];  // [4096, 8, 128, 16]
    const int*   bt   = (const int*)d_in[5];    // [64, 64]
    const int*   ctxl = (const int*)d_in[6];    // [64]
    // d_in[7] (slot_mapping) unused: the scattered slot is exactly token ctx-1
    // of the same sequence, so we substitute knew/vnew in-register instead of
    // mutating the const caches.
    dim3 grid(NUM_KV_, NUM_SEQS_);
    PagedAttention_77034533421648_kernel<<<grid, 128, 0, stream>>>(
        q, knew, vnew, kc, vc, bt, ctxl, (float*)d_out);
}
